// GATAttn_27951647162778
// MI455X (gfx1250) — matrix operand order repaired and checked by reading, not compiled, not run on hardware
//
#include <hip/hip_runtime.h>
#include <hip/hip_bf16.h>

typedef __attribute__((ext_vector_type(16))) _Float16 v16h;
typedef __attribute__((ext_vector_type(8)))  _Float16 h8;
typedef __attribute__((ext_vector_type(8)))  float    v8f;

#define N_NODES 4096
#define F_IN    512
#define F_OUT   64
#define NHEAD   8
#define NEG_BIG (-9.0e15f)
#define L2E     1.4426950408889634f

__device__ __forceinline__ v8f zero8() {
  v8f z = {0.f,0.f,0.f,0.f,0.f,0.f,0.f,0.f};
  return z;
}

__device__ __forceinline__ v8f wmma16(v16h a, v16h b, v8f c) {
  // (neg_a, A, neg_b, B, c_mod, C, reuse_a, reuse_b)
  return __builtin_amdgcn_wmma_f32_16x16x32_f16(false, a, false, b, (short)0, c,
                                                false, false);
}

__device__ __forceinline__ v16h ld2x8(const _Float16* lo, const _Float16* hi) {
  h8 a = *(const h8*)lo;
  h8 b = *(const h8*)hi;
  return __builtin_shufflevector(a, b, 0,1,2,3,4,5,6,7,8,9,10,11,12,13,14,15);
}

// ---------------- kernel 0a: x (f32) -> f16 row-major ----------------
__global__ void __launch_bounds__(256) cvt_x(const float* __restrict__ x,
                                             _Float16* __restrict__ xh, int n) {
  int i = blockIdx.x * blockDim.x + threadIdx.x;
  if (i < n) xh[i] = (_Float16)x[i];
}

// ---------------- kernel 0b: W -> f16 B-operand layout ----------------
// Wb[((h*16 + k/32)*64 + o)*32 + k%32]
__global__ void __launch_bounds__(256) pack_w(const float* __restrict__ W,
                                              _Float16* __restrict__ Wb) {
  int id = blockIdx.x * blockDim.x + threadIdx.x;   // (h*512 + k)*64 + o
  if (id >= NHEAD * F_IN * F_OUT) return;
  int o = id & 63;
  int k = (id >> 6) & 511;
  int h = id >> 15;
  Wb[(((size_t)h * 16 + (k >> 5)) * 64 + o) * 32 + (k & 31)] = (_Float16)W[id];
}

// ---------------- kernel 1: Wh = x @ W[h] (f16 WMMA, f32 accum) -------
// Stores Wh in f16 B-layout (for GEMM2) and computes s_src/s_tgt.
__global__ void __launch_bounds__(128) gemm1(const _Float16* __restrict__ xh,
                                             const _Float16* __restrict__ Wb,
                                             const float* __restrict__ aS,
                                             const float* __restrict__ aT,
                                             _Float16* __restrict__ Whb,
                                             float* __restrict__ s_src,
                                             float* __restrict__ s_tgt) {
  const int lane = threadIdx.x & 31;
  const int wv   = threadIdx.x >> 5;
  const int h    = blockIdx.y;
  const int i0   = (blockIdx.x * 4 + wv) * 16;
  const int lm = lane & 15, hi = lane >> 4;

  v8f c0 = zero8(), c1 = zero8(), c2 = zero8(), c3 = zero8();

  // A operand: lane holds row i0+lm; K = hi*16..hi*16+15 contiguous
  const _Float16* xrow  = xh + (size_t)(i0 + lm) * F_IN + hi * 16;
  // B operand: lane holds col n=lm(+16*nt); K = hi*16..hi*16+15 contiguous
  const _Float16* wbase = Wb + ((size_t)h * 16 * 64) * 32 + (size_t)lm * 32 + hi * 16;

  #pragma unroll
  for (int kt = 0; kt < 16; ++kt) {
    v16h A = ld2x8(xrow + kt * 32, xrow + kt * 32 + 8);
    const _Float16* wk = wbase + (size_t)kt * 64 * 32;
    c0 = wmma16(A, *(const v16h*)(wk + 0 * 16 * 32), c0);
    c1 = wmma16(A, *(const v16h*)(wk + 1 * 16 * 32), c1);
    c2 = wmma16(A, *(const v16h*)(wk + 2 * 16 * 32), c2);
    c3 = wmma16(A, *(const v16h*)(wk + 3 * 16 * 32), c3);
  }

  float aSr[4], aTr[4];
  #pragma unroll
  for (int nt = 0; nt < 4; ++nt) {
    aSr[nt] = aS[h * 64 + nt * 16 + lm];
    aTr[nt] = aT[h * 64 + nt * 16 + lm];
  }

  const v8f cc[4] = {c0, c1, c2, c3};
  float psrc[8], ptgt[8];
  #pragma unroll
  for (int v = 0; v < 8; ++v) {
    // C layout: VGPR v holds row M=v (lanes 0-15) / M=v+8 (lanes 16-31)
    int r = i0 + v + hi * 8;
    float ps = 0.f, pt = 0.f;
    #pragma unroll
    for (int nt = 0; nt < 4; ++nt) {
      float val = cc[nt][v];
      ps += aSr[nt] * val;
      pt += aTr[nt] * val;
      Whb[(((size_t)h * 128 + (r >> 5)) * 64 + nt * 16 + lm) * 32 + (r & 31)] =
          (_Float16)val;
    }
    #pragma unroll
    for (int mk = 8; mk >= 1; mk >>= 1) {
      ps += __shfl_xor(ps, mk, 32);
      pt += __shfl_xor(pt, mk, 32);
    }
    psrc[v] = ps; ptgt[v] = pt;
  }
  if (lm == 0) {
    #pragma unroll
    for (int v = 0; v < 8; ++v) {
      int r = i0 + v + hi * 8;
      s_src[r * 8 + h] = psrc[v];
      s_tgt[r * 8 + h] = ptgt[v];
    }
  }
}

// ---------------- kernel 2: per-row online (max, sum-exp) ------------
__global__ void __launch_bounds__(256) rowstats(const int* __restrict__ adj,
                                                const float* __restrict__ s_src,
                                                const float* __restrict__ s_tgt,
                                                float* __restrict__ m_t,
                                                float* __restrict__ z_t) {
  const int lane = threadIdx.x & 31;
  const int wv   = threadIdx.x >> 5;
  const int i    = blockIdx.x * 8 + wv;

  float sI[NHEAD];
  #pragma unroll
  for (int h = 0; h < NHEAD; ++h) sI[h] = s_src[i * 8 + h];

  float m[NHEAD], s[NHEAD];
  #pragma unroll
  for (int h = 0; h < NHEAD; ++h) { m[h] = -1.8e16f; s[h] = 0.f; }

  const int* arow = adj + (size_t)i * N_NODES;
  for (int j0 = 0; j0 < N_NODES; j0 += 32) {
    int jj = j0 + lane;
    int a  = arow[jj];
    const float* st = s_tgt + (size_t)jj * 8;
    float4 t0 = *(const float4*)st;
    float4 t1 = *(const float4*)(st + 4);
    float sT[8] = {t0.x, t0.y, t0.z, t0.w, t1.x, t1.y, t1.z, t1.w};
    #pragma unroll
    for (int h = 0; h < NHEAD; ++h) {
      float e = sI[h] + sT[h];
      e = fmaxf(e, 0.2f * e);                 // leaky relu
      e = (a > 0) ? e : NEG_BIG;              // adjacency mask
      float mn = fmaxf(m[h], e);
      s[h] = s[h] * exp2f((m[h] - mn) * L2E) + exp2f((e - mn) * L2E);
      m[h] = mn;
    }
  }
  #pragma unroll
  for (int mk = 16; mk >= 1; mk >>= 1) {
    #pragma unroll
    for (int h = 0; h < NHEAD; ++h) {
      float m2 = __shfl_xor(m[h], mk, 32);
      float s2 = __shfl_xor(s[h], mk, 32);
      float mn = fmaxf(m[h], m2);
      s[h] = s[h] * exp2f((m[h] - mn) * L2E) + s2 * exp2f((m2 - mn) * L2E);
      m[h] = mn;
    }
  }
  if (lane == 0) {
    #pragma unroll
    for (int h = 0; h < NHEAD; ++h) {
      m_t[i * 8 + h] = m[h];
      z_t[i * 8 + h] = s[h];
    }
  }
}

// ---------------- kernel 3: out = softmax(e) @ Wh (f16 WMMA) ----------
// 256 threads = 8 waves = 8 heads sharing one 16-row tile; adj staged once.
__global__ void __launch_bounds__(256) gat_agg(const int* __restrict__ adj,
                                               const _Float16* __restrict__ Whb,
                                               const float* __restrict__ s_src,
                                               const float* __restrict__ s_tgt,
                                               const float* __restrict__ m_t,
                                               const float* __restrict__ z_t,
                                               float* __restrict__ out) {
  __shared__ int   adjT[16 * 33];   // padded: stride 33 kills bank conflicts
  __shared__ float stg[32 * 8];     // s_tgt tile, [kk][head]

  const int t    = threadIdx.x;
  const int lane = t & 31;
  const int w    = t >> 5;          // head
  const int lm = lane & 15, hi = lane >> 4;
  const int i0 = blockIdx.x * 16;

  const float sI = s_src[(i0 + lm) * 8 + w];
  const float mI = m_t[(i0 + lm) * 8 + w];

  v8f c0 = zero8(), c1 = zero8(), c2 = zero8(), c3 = zero8();

  for (int j0 = 0; j0 < N_NODES; j0 += 32) {
    {
      int f = t, r = f >> 5, cj = f & 31;
      adjT[r * 33 + cj] = adj[(size_t)(i0 + r) * N_NODES + j0 + cj];
      f = t + 256; r = f >> 5; cj = f & 31;
      adjT[r * 33 + cj] = adj[(size_t)(i0 + r) * N_NODES + j0 + cj];
      stg[t] = s_tgt[(size_t)(j0 + (t >> 3)) * 8 + (t & 7)];
    }
    __syncthreads();

    // Build unnormalized probability tile as the WMMA A operand, in this file's contiguous k order (matches Whb):
    // lane row M = lm; elements 0..15 -> K = hi*16+idx
    v16h A;
    #pragma unroll
    for (int idx = 0; idx < 16; ++idx) {
      int kk = hi * 16 + idx;
      float e = sI + stg[kk * 8 + w];
      e = fmaxf(e, 0.2f * e);
      e = (adjT[lm * 33 + kk] > 0) ? e : NEG_BIG;
      A[idx] = (_Float16)exp2f((e - mI) * L2E);
    }
    __syncthreads();

    const _Float16* wb =
        Whb + (((size_t)w * 128 + (j0 >> 5)) * 64 + lm) * 32 + hi * 16;
    c0 = wmma16(A, *(const v16h*)(wb + 0 * 16 * 32), c0);
    c1 = wmma16(A, *(const v16h*)(wb + 1 * 16 * 32), c1);
    c2 = wmma16(A, *(const v16h*)(wb + 2 * 16 * 32), c2);
    c3 = wmma16(A, *(const v16h*)(wb + 3 * 16 * 32), c3);
  }

  const v8f cc[4] = {c0, c1, c2, c3};
  #pragma unroll
  for (int v = 0; v < 8; ++v) {
    int r = i0 + v + hi * 8;
    float inv = 1.0f / z_t[r * 8 + w];
    #pragma unroll
    for (int nt = 0; nt < 4; ++nt) {
      out[((size_t)w * N_NODES + r) * F_OUT + nt * 16 + lm] = cc[nt][v] * inv;
    }
  }
}

extern "C" void kernel_launch(void* const* d_in, const int* in_sizes, int n_in,
                              void* d_out, int out_size, void* d_ws, size_t ws_size,
                              hipStream_t stream) {
  const float* x   = (const float*)d_in[0];
  const int*   adj = (const int*)d_in[1];
  const float* W   = (const float*)d_in[2];
  const float* aS  = (const float*)d_in[3];
  const float* aT  = (const float*)d_in[4];
  float* out = (float*)d_out;

  char* ws = (char*)d_ws;
  _Float16* xh  = (_Float16*)ws;                                   // 4 MB
  _Float16* Wb  = (_Float16*)(ws + (4u << 20));                    // 512 KB
  _Float16* Whb = (_Float16*)(ws + (4u << 20) + (512u << 10));     // 4 MB
  float* s_src  = (float*)(ws + (8u << 20) + (512u << 10));        // 128 KB
  float* s_tgt  = s_src + N_NODES * NHEAD;
  float* m_t    = s_tgt + N_NODES * NHEAD;
  float* z_t    = m_t + N_NODES * NHEAD;

  cvt_x<<<(N_NODES * F_IN) / 256, 256, 0, stream>>>(x, xh, N_NODES * F_IN);
  pack_w<<<(NHEAD * F_IN * F_OUT) / 256, 256, 0, stream>>>(W, Wb);
  gemm1<<<dim3(64, NHEAD), 128, 0, stream>>>(xh, Wb, aS, aT, Whb, s_src, s_tgt);
  rowstats<<<N_NODES / 8, 256, 0, stream>>>(adj, s_src, s_tgt, m_t, z_t);
  gat_agg<<<N_NODES / 16, 256, 0, stream>>>(adj, Whb, s_src, s_tgt, m_t, z_t, out);
}
